// GemNetT_24163486008150
// MI455X (gfx1250) — matrix operand order repaired and checked by reading, not compiled, not run on hardware
//
#include <hip/hip_runtime.h>

// ---------------------------------------------------------------------------
// GemNet-T for MI455X (gfx1250, wave32, WMMA).
// All dense layers -> one bf16 WMMA GEMM (v_wmma_f32_16x16x32_bf16, f32 acc,
// fused bias/ScaledSiLU/residual epilogue). LDS tiles are stored in WMMA
// fragment order so fragment loads are ds_load_b128 (2 per fragment) instead
// of scalar ds_load_u16. Weights are repacked into B-fragment order during
// the per-launch f32->bf16 conversion. Network is HBM-bound (~6 GB traffic
// @ 23.3 TB/s vs ~0.9 TFLOP), so bf16 storage + matrix pipes is the roofline
// choice. Irregular ops (segment sums, triplet scatter, RBF, Legendre) are
// f32 VALU + global f32 atomics.
// ---------------------------------------------------------------------------

#define DEV_INLINE __device__ __forceinline__

typedef unsigned short bfraw; // raw bf16 bits
typedef __attribute__((ext_vector_type(16))) __bf16 v16bf;
typedef __attribute__((ext_vector_type(8)))  float  v8f;

DEV_INLINE float bf2f(bfraw h) {
  union { unsigned u; float f; } v; v.u = ((unsigned)h) << 16; return v.f;
}
DEV_INLINE bfraw f2bf(float f) {
  union { float f; unsigned u; } v; v.f = f;
  unsigned r = v.u + 0x7FFFu + ((v.u >> 16) & 1u); // round-to-nearest-even
  return (bfraw)(r >> 16);
}
DEV_INLINE float dssilu(float x) { // GemNet ScaledSiLU = silu(x)/0.6
  return x * (1.0f / (1.0f + __expf(-x))) * (1.0f / 0.6f);
}

#define INV_SQRT2F 0.7071067811865476f

// ---------------------------------------------------------------------------
// WMMA GEMM: C[M,N] = epilogue(A[M,K] @ B[K,N])
//   A: bf16 row-major (K multiple of 16, 32B-aligned rows-chunks)
//   Bf: bf16 pre-repacked into WMMA-B fragment order:
//       Bf[ (kb*NB + tc)*512 + lane*16 + idx ]
//       lane = (n&15) + 16*((k&31)>>4), idx = k&15, kb = k>>5, tc = n>>4
//   epilogue: (+bias[n]) -> (ssilu if act) -> ((skip + y)*skip_scale if skip)
// Block = 256 threads = 8 waves (4x2), tile 128x128, K step 32.
// LDS fragment layout: seg[tile(8)][lane(32)][16] -> lane reads 32B contig.
// ---------------------------------------------------------------------------
__global__ __launch_bounds__(256) void k_gemm(
    const bfraw* __restrict__ A, const bfraw* __restrict__ Bf,
    bfraw* __restrict__ Cb, float* __restrict__ Cf,
    const float* __restrict__ bias, const bfraw* __restrict__ skip,
    int M, int N, int K, int act, float skip_scale)
{
  __shared__ bfraw As_f[8 * 32 * 16];  // 8KB, A fragments
  __shared__ bfraw Bs_f[8 * 32 * 16];  // 8KB, B fragments
  const int tid  = threadIdx.x;
  const int lane = tid & 31;
  const int w    = tid >> 5;
  const int wm   = w >> 1, wn = w & 1;      // 4x2 wave grid
  const long bm  = (long)blockIdx.y * 128;
  const long bn  = (long)blockIdx.x * 128;
  const int  NB  = (N + 15) >> 4;

  v8f acc[2][4];
#pragma unroll
  for (int i = 0; i < 2; ++i)
#pragma unroll
    for (int j = 0; j < 4; ++j)
#pragma unroll
      for (int v = 0; v < 8; ++v) acc[i][j][v] = 0.0f;

  // A loader: thread -> (row ar, k-half ah). The 16 contiguous global bf16
  // go to one lane (k<16 -> lane mr, k>=16 -> mr+16), idx = k&15:
  // matches the contiguous-k B-fragment layout.
  const int ar     = tid >> 1;
  const int ah     = (tid & 1) << 4;
  const int tile_a = ar >> 4, mr = ar & 15;
  const int idxb   = (ah >> 4) << 3;
  bfraw* dstA0 = &As_f[(((tile_a << 5) + mr + ah) << 4)];
  bfraw* dstA1 = &As_f[(((tile_a << 5) + mr + ah) << 4) + 8];
  // B loader: pure contiguous copy of one 8KB fragment-ordered region.
  const int btc   = tid >> 5;         // local tilecol 0..7
  const int bchnk = (tid & 31) << 4;  // 16-ushort chunk within tilecol
  bfraw* dstB = &Bs_f[(btc << 9) + bchnk];

  const uint4 zq = make_uint4(0u, 0u, 0u, 0u);
  for (int k0 = 0; k0 < K; k0 += 32) {
    const long gr = bm + ar;
    uint4 a0 = zq, a1 = zq;
    if (gr < M && (k0 + ah) < K) {
      const uint4* gp = (const uint4*)(A + gr * (long)K + k0 + ah);
      if (k0 + 32 < K) __builtin_prefetch(gp + 4, 0, 1); // next K tile
      a0 = gp[0]; a1 = gp[1];
    }
    ((uint4*)dstA0)[0] = a0;
    ((uint4*)dstA1)[0] = a1;

    const int kb = k0 >> 5;
    const int tc = (int)(bn >> 4) + btc;
    uint4 b0 = zq, b1 = zq;
    if (tc < NB) {
      const uint4* gq =
          (const uint4*)(Bf + (((long)kb * NB + tc) << 9) + bchnk);
      b0 = gq[0]; b1 = gq[1];
    }
    ((uint4*)dstB)[0] = b0;
    ((uint4*)dstB)[1] = b1;
    __syncthreads();

    union FragU { v16bf v; uint4 q[2]; } af[2], bfr[4];
#pragma unroll
    for (int i = 0; i < 2; ++i) {
      const int base = ((((wm << 1) + i) << 5) + lane) << 4;
      af[i].q[0] = *(const uint4*)&As_f[base];
      af[i].q[1] = *(const uint4*)&As_f[base + 8];
    }
#pragma unroll
    for (int j = 0; j < 4; ++j) {
      const int base = ((((wn << 2) + j) << 5) + lane) << 4;
      bfr[j].q[0] = *(const uint4*)&Bs_f[base];
      bfr[j].q[1] = *(const uint4*)&Bs_f[base + 8];
    }
#pragma unroll
    for (int i = 0; i < 2; ++i)
#pragma unroll
      for (int j = 0; j < 4; ++j)
        acc[i][j] = __builtin_amdgcn_wmma_f32_16x16x32_bf16(
            false, af[i].v, false, bfr[j].v, (short)0, acc[i][j], false, false);
    __syncthreads();
  }

  // C/D layout: lane col = lane&15; row m = v + (lane<16 ? 0 : 8)
  const int roff = (lane < 16) ? 0 : 8;
  const int ncl  = lane & 15;
#pragma unroll
  for (int i = 0; i < 2; ++i)
#pragma unroll
    for (int j = 0; j < 4; ++j)
#pragma unroll
      for (int v = 0; v < 8; ++v) {
        const long gm = bm + wm * 32 + i * 16 + v + roff;
        const long gn = bn + wn * 64 + j * 16 + ncl;
        if (gm < M && gn < N) {
          float x = acc[i][j][v];
          if (bias) x += bias[gn];
          if (act)  x = dssilu(x);
          const long o = gm * N + gn;
          if (skip) x = (bf2f(skip[o]) + x) * skip_scale;
          if (Cb) Cb[o] = f2bf(x);
          if (Cf) Cf[o] = x;
        }
      }
}

// ---------------------------------------------------------------------------
// Weight repack: f32 row-major (K,N) -> bf16 WMMA-B fragment order.
// Dest-driven so partially-filled k-blocks (K%32==16) are zero-padded.
// ---------------------------------------------------------------------------
__global__ void k_repackB(bfraw* o, const float* W, int K, int N) {
  const int NB = (N + 15) >> 4;
  const int KB = (K + 31) >> 5;
  long g = blockIdx.x * 256L + threadIdx.x;
  if (g >= (long)KB * NB * 512) return;
  int idx = (int)(g & 15);
  int l   = (int)((g >> 4) & 31);
  int tc  = (int)((g >> 9) % NB);
  int kb  = (int)(g / ((long)NB << 9));
  int n = (tc << 4) + (l & 15);
  int k = (kb << 5) + ((l >> 4) << 4) + idx;
  o[g] = (k < K && n < N) ? f2bf(W[(long)k * N + n]) : (bfraw)0;
}
// W_cbf3_down (S=7,R=128,EC=16): logical B is (K=128, N=112) with
// B[k, s*16+i] = W[(s*128+k)*16+i]; emitted directly in fragment order.
__global__ void k_repack_cbf_B(bfraw* o, const float* W) {
  const int NB = 7, KB = 4;
  long g = blockIdx.x * 256L + threadIdx.x;
  if (g >= (long)KB * NB * 512) return;
  int idx = (int)(g & 15);
  int l   = (int)((g >> 4) & 31);
  int tc  = (int)((g >> 9) % NB);          // = s
  int kb  = (int)(g / ((long)NB << 9));
  int i = l & 15;
  int k = (kb << 5) + ((l >> 4) << 4) + idx;
  o[g] = (k < 128) ? f2bf(W[((long)tc * 128 + k) * 16 + i]) : (bfraw)0;
}

// ---------------------------------------------------------------------------
// Elementwise / irregular kernels
// ---------------------------------------------------------------------------
__global__ void k_zero_f32(float* p, long n) {
  long i = blockIdx.x * 256L + threadIdx.x;
  if (i < n) p[i] = 0.0f;
}
__global__ void k_f32_to_bf16(bfraw* d, const float* s, long n) {
  long i = blockIdx.x * 256L + threadIdx.x;
  if (i < n) d[i] = f2bf(s[i]);
}
__global__ void k_mul_bf16(bfraw* a, const bfraw* g, long n) { // a *= g
  long i = blockIdx.x * 256L + threadIdx.x;
  if (i < n) a[i] = f2bf(bf2f(a[i]) * bf2f(g[i]));
}
__global__ void k_add_scale(bfraw* o, const bfraw* a, const bfraw* b, float s, long n) {
  long i = blockIdx.x * 256L + threadIdx.x;
  if (i < n) o[i] = f2bf((bf2f(a[i]) + bf2f(b[i])) * s);
}
// out[e,c] = (a[e,c] + b[swap[e],c]) * s   (out may alias a)
__global__ void k_swap_add(bfraw* o, const bfraw* a, const bfraw* b,
                           const int* sw, float s, int E, int C) {
  long i = blockIdx.x * 256L + threadIdx.x;
  if (i >= (long)E * C) return;
  int e = (int)(i / C), c = (int)(i % C);
  o[i] = f2bf((bf2f(a[i]) + bf2f(b[(long)sw[e] * C + c])) * s);
}
// acc[idx[e],c] += x[e,c] * g[e,c]  (f32 atomics)
__global__ void k_scatter_mulgate(float* acc, const bfraw* x, const bfraw* g,
                                  const int* idx, int E, int C) {
  long i = blockIdx.x * 256L + threadIdx.x;
  if (i >= (long)E * C) return;
  int e = (int)(i / C), c = (int)(i % C);
  atomicAdd(&acc[(long)idx[e] * C + c], bf2f(x[i]) * bf2f(g[i]));
}
// out[e] = [h[idx_s[e]] (HC), h[idx_t[e]] (HC), third[e] (TC)]
__global__ void k_concat3(bfraw* o, const bfraw* h, const int* is, const int* it,
                          const bfraw* third, int E, int HC, int TC) {
  const int W = 2 * HC + TC;
  long i = blockIdx.x * 256L + threadIdx.x;
  if (i >= (long)E * W) return;
  int e = (int)(i / W), j = (int)(i % W);
  bfraw v;
  if (j < HC)            v = h[(long)is[e] * HC + j];
  else if (j < 2 * HC)   v = h[(long)it[e] * HC + (j - HC)];
  else                   v = third[(long)e * TC + (j - 2 * HC)];
  o[i] = v;
}
// h0[a] = [atom_emb[type[a]] (512), z[batch[a]] (256)]  -> bf16
__global__ void k_build_h0(bfraw* o, const float* emb, const int* types,
                           const float* z, const int* batch, int N) {
  long i = blockIdx.x * 256L + threadIdx.x;
  if (i >= (long)N * 768) return;
  int a = (int)(i / 768), j = (int)(i % 768);
  float v = (j < 512) ? emb[(long)types[a] * 512 + j]
                      : z[(long)batch[a] * 256 + (j - 512)];
  o[i] = f2bf(v);
}
__global__ void k_normalize(float* Vn, const float* V, int E) {
  int e = blockIdx.x * 256 + threadIdx.x;
  if (e >= E) return;
  float x = V[e * 3], y = V[e * 3 + 1], z = V[e * 3 + 2];
  float r = rsqrtf(x * x + y * y + z * z);
  Vn[e * 3] = x * r; Vn[e * 3 + 1] = y * r; Vn[e * 3 + 2] = z * r;
}
// Gaussian RBF * poly envelope (p=6): out (E,128) bf16
__global__ void k_rbf(bfraw* o, const float* D, int E) {
  long i = blockIdx.x * 256L + threadIdx.x;
  if (i >= (long)E * 128) return;
  int e = (int)(i >> 7), k = (int)(i & 127);
  float d = D[e] * (1.0f / 6.0f);
  float d2 = d * d, d6 = d2 * d2 * d2;
  float env = 1.0f - 28.0f * d6 + 48.0f * d6 * d - 21.0f * d6 * d2;
  env = (d < 1.0f) ? env : 0.0f;
  float t = d - (float)k * (1.0f / 127.0f);
  o[i] = f2bf(env * __expf(-8064.5f * t * t)); // -0.5/(1/127)^2
}
// zero-m spherical harmonics via Legendre recurrence: sph (T,7) f32
__global__ void k_sph(float* sph, const float* Vn, const int* ba, const int* ca, int T) {
  long t = blockIdx.x * 256L + threadIdx.x;
  if (t >= T) return;
  int eb = ba[t], ec = ca[t];
  float c = Vn[eb * 3] * Vn[ec * 3] + Vn[eb * 3 + 1] * Vn[ec * 3 + 1] +
            Vn[eb * 3 + 2] * Vn[ec * 3 + 2];
  c = fminf(1.0f, fmaxf(-1.0f, c));
  float P[7];
  P[0] = 1.0f; P[1] = c;
#pragma unroll
  for (int l = 2; l < 7; ++l)
    P[l] = ((2 * l - 1) * c * P[l - 1] - (l - 1) * P[l - 2]) / (float)l;
#pragma unroll
  for (int l = 0; l < 7; ++l)
    sph[t * 7 + l] = P[l] * sqrtf((2.0f * l + 1.0f) / 12.566370614359172f);
}
// sumk[ca[t], s, c] += xb[ba[t], c] * sph[t, s]   (E,7,64) f32
__global__ void k_triplet_scatter(float* sumk, const bfraw* xb, const float* sph,
                                  const int* ba, const int* ca, int T) {
  long i = blockIdx.x * 256L + threadIdx.x;
  if (i >= (long)T * 64) return;
  int t = (int)(i >> 6), c = (int)(i & 63);
  float v = bf2f(xb[(long)ba[t] * 64 + c]);
  long base = ((long)ca[t] * 7) * 64 + c;
#pragma unroll
  for (int s = 0; s < 7; ++s)
    atomicAdd(&sumk[base + (long)s * 64], v * sph[(long)t * 7 + s]);
}
// tmp[e, c*16+i] = sum_s rbfW1[e, s*16+i] * sumk[e, s, c]   (E,1024) bf16
__global__ void k_bil_tmp(bfraw* tmp, const bfraw* rbfW1, const float* sumk, int E) {
  long g = blockIdx.x * 256L + threadIdx.x;
  if (g >= (long)E * 1024) return;
  int e = (int)(g >> 10), k = (int)(g & 1023);
  int c = k >> 4, i = k & 15;
  float acc = 0.0f;
#pragma unroll
  for (int s = 0; s < 7; ++s)
    acc += bf2f(rbfW1[(long)e * 112 + s * 16 + i]) *
           sumk[((long)e * 7 + s) * 64 + c];
  tmp[g] = f2bf(acc);
}
// out[m] += dot(A[m,:] bf16, w f32)   one wave per row
__global__ void k_matvec_acc(float* out, const bfraw* A, const float* w, int M, int K) {
  int row  = (int)((blockIdx.x * 256L + threadIdx.x) >> 5);
  int lane = threadIdx.x & 31;
  if (row >= M) return;
  float s = 0.0f;
  for (int k = lane; k < K; k += 32) s += bf2f(A[(long)row * K + k]) * w[k];
#pragma unroll
  for (int o = 16; o > 0; o >>= 1) s += __shfl_down(s, o, 32);
  if (lane == 0) out[row] += s;
}
__global__ void k_final_energy(float* out, const float* Ea, const int* batch, int N) {
  int a = blockIdx.x * 256 + threadIdx.x;
  if (a < N) atomicAdd(&out[batch[a]], Ea[a]);
}
__global__ void k_final_forces(float* out, const float* Fe, const float* Vn,
                               const int* idxt, int E) {
  int e = blockIdx.x * 256 + threadIdx.x;
  if (e >= E) return;
  float f = Fe[e];
#pragma unroll
  for (int j = 0; j < 3; ++j)
    atomicAdd(&out[(long)idxt[e] * 3 + j], f * Vn[e * 3 + j]);
}

// ---------------------------------------------------------------------------
// Host orchestration
// ---------------------------------------------------------------------------
extern "C" void kernel_launch(void* const* d_in, const int* in_sizes, int n_in,
                              void* d_out, int out_size, void* d_ws, size_t ws_size,
                              hipStream_t stream) {
  (void)in_sizes; (void)n_in; (void)out_size; (void)ws_size;
  const int Na = 1024, Ea = 24576, Ta = 294912, Ba = 32;
  const int EA = 512, EE = 512, ET = 64, ER = 16, EB = 64;
  const int Rr = 128, ZD = 256, NT = 1, NBLK = 3;

  // ---- input index map: params dict flattened in insertion order ----
  const int IP_W_RBF3 = 0, IP_W_CBF3 = 1, IP_W_RBF_H = 2, IP_W_RBF_OUT = 3,
            IP_ATOM_EMB = 4, IP_W_LAT = 5, IP_B_LAT = 6, IP_W_EDGE = 7;
  const int IP_INT0 = 8;            // +0 dense_ca,+1 dense_ba,+2 mlp_rbf,+3 down,
                                    // +4 Wbil,+5 up_ca,+6 up_ac,+7..8 res_before,
                                    // +9..12 res_after,+13 au_rbf,+14 au0,
                                    // +15..20 au_res,+21 concat,+22..23 res_m
  const int IP_OUT0 = IP_INT0 + 3 * 24; // 80: +0 dense_rbf,+1 layer0,+2..7 res_E,
                                        // +8 out_energy,+9 f0,+10..15 res_F,
                                        // +16 dense_rbf_F,+17 out_forces
  const int IP_Z = IP_OUT0 + 4 * 18;    // 152
  const int IP_VST = IP_Z + 1, IP_DST = IP_Z + 2, IP_ATYPE = IP_Z + 3,
            IP_BATCH = IP_Z + 4, IP_IDXS = IP_Z + 5, IP_IDXT = IP_Z + 6,
            IP_ID3BA = IP_Z + 7, IP_ID3CA = IP_Z + 8, IP_IDSWAP = IP_Z + 9;

  auto F32 = [&](int i) { return (const float*)d_in[i]; };
  auto I32 = [&](int i) { return (const int*)d_in[i]; };

  // bump allocator over workspace (same deterministic layout each launch)
  char* wp = (char*)d_ws;
  auto alloc = [&](size_t bytes) {
    void* r = (void*)wp; wp += (bytes + 255) & ~(size_t)255; return r;
  };
  auto nblk = [](long n) { return (unsigned)((n + 255) / 256); };
  auto zero = [&](float* p, long n) {
    k_zero_f32<<<nblk(n), 256, 0, stream>>>(p, n);
  };
  auto cvt = [&](bfraw* d, const float* s, long n) {
    k_f32_to_bf16<<<nblk(n), 256, 0, stream>>>(d, s, n);
  };
  // weight -> bf16 in WMMA-B fragment order
  auto tobfB = [&](const float* s, int K, int N) {
    const int NB = (N + 15) >> 4, KB = (K + 31) >> 5;
    long total = (long)KB * NB * 512;
    bfraw* d = (bfraw*)alloc(total * 2);
    k_repackB<<<nblk(total), 256, 0, stream>>>(d, s, K, N);
    return (const bfraw*)d;
  };
  auto gemm = [&](const bfraw* A, const bfraw* B, bfraw* Cb, float* Cf,
                  const float* bias, const bfraw* skip,
                  int M, int Nn, int K, int act, float ss) {
    dim3 g((Nn + 127) / 128, (M + 127) / 128);
    k_gemm<<<g, 256, 0, stream>>>(A, B, Cb, Cf, bias, skip, M, Nn, K, act, ss);
  };
  // GemNet residual layer: x = (x + ssilu(ssilu(x@W0)@W1)) / sqrt(2)
  auto residual = [&](bfraw* x, const bfraw* W0, const bfraw* W1, bfraw* tmp,
                      int M, int D) {
    gemm(x, W0, tmp, nullptr, nullptr, nullptr, M, D, D, 1, 0.0f);
    gemm(tmp, W1, x, nullptr, nullptr, x, M, D, D, 1, INV_SQRT2F);
  };

  // ---- activation buffers ----
  float* Vn      = (float*)alloc((size_t)Ea * 3 * 4);
  float* sph     = (float*)alloc((size_t)Ta * 7 * 4);
  bfraw* rbf_bf  = (bfraw*)alloc((size_t)Ea * Rr * 2);
  bfraw* rbf3_bf = (bfraw*)alloc((size_t)Ea * ER * 2);
  bfraw* rbfh_bf = (bfraw*)alloc((size_t)Ea * ER * 2);
  bfraw* rbfo_bf = (bfraw*)alloc((size_t)Ea * ER * 2);
  bfraw* rbfW1   = (bfraw*)alloc((size_t)Ea * 112 * 2);
  bfraw* cbf_rp  = (bfraw*)alloc((size_t)4 * 7 * 512 * 2);
  bfraw* h0_bf   = (bfraw*)alloc((size_t)Na * 768 * 2);
  bfraw* h_bf    = (bfraw*)alloc((size_t)Na * EA * 2);
  bfraw* cat_bf  = (bfraw*)alloc((size_t)Ea * 1536 * 2);
  bfraw* m_bf    = (bfraw*)alloc((size_t)Ea * EE * 2);
  bfraw* x_bf    = (bfraw*)alloc((size_t)Ea * EE * 2);
  bfraw* sA      = (bfraw*)alloc((size_t)Ea * EE * 2); // gates
  bfraw* sB      = (bfraw*)alloc((size_t)Ea * EE * 2); // residual tmp
  bfraw* sC      = (bfraw*)alloc((size_t)Ea * EE * 2); // x_skip
  bfraw* sD      = (bfraw*)alloc((size_t)Ea * EE * 2); // xba / m2 / xF
  bfraw* sE      = (bfraw*)alloc((size_t)Ea * EE * 2); // up_ca / x3
  bfraw* sF      = (bfraw*)alloc((size_t)Ea * EE * 2); // up_ac
  bfraw* xb_bf   = (bfraw*)alloc((size_t)Ea * ET * 2);
  bfraw* xbil_bf = (bfraw*)alloc((size_t)Ea * EB * 2);
  bfraw* tmpbil  = (bfraw*)alloc((size_t)Ea * 1024 * 2);
  float* sumk    = (float*)alloc((size_t)Ea * 7 * 64 * 4);
  float* xE_acc  = (float*)alloc((size_t)Na * EE * 4);
  bfraw* xE_bf   = (bfraw*)alloc((size_t)Na * EE * 2);
  bfraw* h2_bf   = (bfraw*)alloc((size_t)Na * EE * 2);
  bfraw* ta_bf   = (bfraw*)alloc((size_t)Na * EA * 2);
  bfraw* tb_bf   = (bfraw*)alloc((size_t)Na * EA * 2);
  float* E_acc   = (float*)alloc((size_t)Na * 4);
  float* F_acc   = (float*)alloc((size_t)Ea * 4);

  // ---- init / basics ----
  zero((float*)d_out, Ba * NT + (long)Na * 3);
  zero(E_acc, Na);
  zero(F_acc, Ea);
  k_normalize<<<nblk(Ea), 256, 0, stream>>>(Vn, F32(IP_VST), Ea);
  k_rbf<<<nblk((long)Ea * Rr), 256, 0, stream>>>(rbf_bf, F32(IP_DST), Ea);
  k_sph<<<nblk(Ta), 256, 0, stream>>>(sph, Vn, I32(IP_ID3BA), I32(IP_ID3CA), Ta);
  k_repack_cbf_B<<<nblk((long)4 * 7 * 512), 256, 0, stream>>>(cbf_rp, F32(IP_W_CBF3));

  gemm(rbf_bf, tobfB(F32(IP_W_RBF3), Rr, ER), rbf3_bf, nullptr, nullptr, nullptr,
       Ea, ER, Rr, 0, 0.0f);
  gemm(rbf_bf, tobfB(F32(IP_W_RBF_H), Rr, ER), rbfh_bf, nullptr, nullptr, nullptr,
       Ea, ER, Rr, 0, 0.0f);
  gemm(rbf_bf, tobfB(F32(IP_W_RBF_OUT), Rr, ER), rbfo_bf, nullptr, nullptr, nullptr,
       Ea, ER, Rr, 0, 0.0f);
  gemm(rbf_bf, cbf_rp, rbfW1, nullptr, nullptr, nullptr, Ea, 112, Rr, 0, 0.0f);

  // ---- atom + edge embeddings ----
  k_build_h0<<<nblk((long)Na * 768), 256, 0, stream>>>(
      h0_bf, F32(IP_ATOM_EMB), I32(IP_ATYPE), F32(IP_Z), I32(IP_BATCH), Na);
  gemm(h0_bf, tobfB(F32(IP_W_LAT), EA + ZD, EA), h_bf, nullptr,
       F32(IP_B_LAT), nullptr, Na, EA, EA + ZD, 0, 0.0f);
  k_concat3<<<nblk((long)Ea * 1152), 256, 0, stream>>>(
      cat_bf, h_bf, I32(IP_IDXS), I32(IP_IDXT), rbf_bf, Ea, EA, Rr);
  gemm(cat_bf, tobfB(F32(IP_W_EDGE), 2 * EA + Rr, EE), m_bf, nullptr,
       nullptr, nullptr, Ea, EE, 2 * EA + Rr, 1, 0.0f);

  // ---- output block ----
  auto out_block = [&](int ob, const bfraw* m) {
    gemm(rbfo_bf, tobfB(F32(ob + 0), ER, EE), sA, nullptr, nullptr, nullptr,
         Ea, EE, ER, 0, 0.0f);
    zero(xE_acc, (long)Na * EE);
    k_scatter_mulgate<<<nblk((long)Ea * EE), 256, 0, stream>>>(
        xE_acc, m, sA, I32(IP_IDXT), Ea, EE);
    cvt(xE_bf, xE_acc, (long)Na * EE);
    gemm(xE_bf, tobfB(F32(ob + 1), EE, EA), ta_bf, nullptr, nullptr, nullptr,
         Na, EA, EE, 1, 0.0f);
    for (int r = 0; r < 3; ++r)
      residual(ta_bf, tobfB(F32(ob + 2 + 2 * r), EA, EA),
               tobfB(F32(ob + 3 + 2 * r), EA, EA), tb_bf, Na, EA);
    k_matvec_acc<<<nblk((long)Na * 32), 256, 0, stream>>>(
        E_acc, ta_bf, F32(ob + 8), Na, EA);
    gemm(m, tobfB(F32(ob + 9), EE, EE), sD, nullptr, nullptr, nullptr,
         Ea, EE, EE, 1, 0.0f);
    for (int r = 0; r < 3; ++r)
      residual(sD, tobfB(F32(ob + 10 + 2 * r), EE, EE),
               tobfB(F32(ob + 11 + 2 * r), EE, EE), sB, Ea, EE);
    gemm(rbfo_bf, tobfB(F32(ob + 16), ER, EE), sA, nullptr, nullptr, nullptr,
         Ea, EE, ER, 0, 0.0f);
    k_mul_bf16<<<nblk((long)Ea * EE), 256, 0, stream>>>(sD, sA, (long)Ea * EE);
    k_matvec_acc<<<nblk((long)Ea * 32), 256, 0, stream>>>(
        F_acc, sD, F32(ob + 17), Ea, EE);
  };

  out_block(IP_OUT0, m_bf);

  // ---- interaction blocks ----
  for (int i = 0; i < NBLK; ++i) {
    const int ib = IP_INT0 + i * 24;
    gemm(m_bf, tobfB(F32(ib + 0), EE, EE), sC, nullptr, nullptr, nullptr,
         Ea, EE, EE, 1, 0.0f);                                   // x_skip
    gemm(m_bf, tobfB(F32(ib + 1), EE, EE), sD, nullptr, nullptr, nullptr,
         Ea, EE, EE, 1, 0.0f);                                   // dense_ba
    gemm(rbf3_bf, tobfB(F32(ib + 2), ER, EE), sA, nullptr, nullptr, nullptr,
         Ea, EE, ER, 0, 0.0f);                                   // mlp_rbf gate
    k_mul_bf16<<<nblk((long)Ea * EE), 256, 0, stream>>>(sD, sA, (long)Ea * EE);
    gemm(sD, tobfB(F32(ib + 3), EE, ET), xb_bf, nullptr, nullptr, nullptr,
         Ea, ET, EE, 1, 0.0f);                                   // down -> (E,64)
    zero(sumk, (long)Ea * 7 * 64);
    k_triplet_scatter<<<nblk((long)Ta * ET), 256, 0, stream>>>(
        sumk, xb_bf, sph, I32(IP_ID3BA), I32(IP_ID3CA), Ta);
    k_bil_tmp<<<nblk((long)Ea * 1024), 256, 0, stream>>>(tmpbil, rbfW1, sumk, Ea);
    gemm(tmpbil, tobfB(F32(ib + 4), 1024, EB), xbil_bf, nullptr, nullptr, nullptr,
         Ea, EB, 1024, 0, 0.0f);
    gemm(xbil_bf, tobfB(F32(ib + 5), EB, EE), sE, nullptr, nullptr, nullptr,
         Ea, EE, EB, 1, 0.0f);                                   // up_ca
    gemm(xbil_bf, tobfB(F32(ib + 6), EB, EE), sF, nullptr, nullptr, nullptr,
         Ea, EE, EB, 1, 0.0f);                                   // up_ac
    k_swap_add<<<nblk((long)Ea * EE), 256, 0, stream>>>(
        sE, sE, sF, I32(IP_IDSWAP), INV_SQRT2F, Ea, EE);         // x3
    k_add_scale<<<nblk((long)Ea * EE), 256, 0, stream>>>(
        x_bf, sC, sE, INV_SQRT2F, (long)Ea * EE);                // (skip+x3)/sqrt2
    residual(x_bf, tobfB(F32(ib + 7), EE, EE), tobfB(F32(ib + 8), EE, EE),
             sB, Ea, EE);                                        // res_before
    k_add_scale<<<nblk((long)Ea * EE), 256, 0, stream>>>(
        x_bf, x_bf, m_bf, INV_SQRT2F, (long)Ea * EE);            // (x+m)/sqrt2
    for (int r = 0; r < 2; ++r)                                  // res_after
      residual(x_bf, tobfB(F32(ib + 9 + 2 * r), EE, EE),
               tobfB(F32(ib + 10 + 2 * r), EE, EE), sB, Ea, EE);
    // atom update
    gemm(rbfh_bf, tobfB(F32(ib + 13), ER, EE), sA, nullptr, nullptr, nullptr,
         Ea, EE, ER, 0, 0.0f);
    zero(xE_acc, (long)Na * EE);
    k_scatter_mulgate<<<nblk((long)Ea * EE), 256, 0, stream>>>(
        xE_acc, x_bf, sA, I32(IP_IDXT), Ea, EE);
    cvt(h2_bf, xE_acc, (long)Na * EE);
    gemm(h2_bf, tobfB(F32(ib + 14), EE, EA), ta_bf, nullptr, nullptr, nullptr,
         Na, EA, EE, 1, 0.0f);
    for (int r = 0; r < 3; ++r)                                  // au_res
      residual(ta_bf, tobfB(F32(ib + 15 + 2 * r), EA, EA),
               tobfB(F32(ib + 16 + 2 * r), EA, EA), tb_bf, Na, EA);
    k_add_scale<<<nblk((long)Na * EA), 256, 0, stream>>>(
        h_bf, h_bf, ta_bf, INV_SQRT2F, (long)Na * EA);
    // concat / edge refresh
    k_concat3<<<nblk((long)Ea * 1536), 256, 0, stream>>>(
        cat_bf, h_bf, I32(IP_IDXS), I32(IP_IDXT), x_bf, Ea, EA, EE);
    gemm(cat_bf, tobfB(F32(ib + 21), 2 * EA + EE, EE), sD, nullptr,
         nullptr, nullptr, Ea, EE, 2 * EA + EE, 1, 0.0f);        // m2
    residual(sD, tobfB(F32(ib + 22), EE, EE), tobfB(F32(ib + 23), EE, EE),
             sB, Ea, EE);                                        // res_m
    k_add_scale<<<nblk((long)Ea * EE), 256, 0, stream>>>(
        m_bf, sD, x_bf, INV_SQRT2F, (long)Ea * EE);              // m=(m2+x)/sqrt2
    out_block(IP_OUT0 + (i + 1) * 18, m_bf);
  }

  // ---- final reductions into d_out: [E_g (B*NT=32) | F_at (N*3)] ----
  float* out_f = (float*)d_out;
  k_final_energy<<<nblk(Na), 256, 0, stream>>>(out_f, E_acc, I32(IP_BATCH), Na);
  k_final_forces<<<nblk(Ea), 256, 0, stream>>>(out_f + Ba * NT, F_acc, Vn,
                                               I32(IP_IDXT), Ea);
}
